// SACRSN_v84_88381837017747
// MI455X (gfx1250) — matrix operand order repaired and checked by reading, not compiled, not run on hardware
//
#include <hip/hip_runtime.h>
#include <math.h>
#include <stdint.h>

typedef __attribute__((ext_vector_type(16))) _Float16 v16h;
typedef __attribute__((ext_vector_type(8)))  float    v8f;
typedef __attribute__((ext_vector_type(2)))  float    v2f;
typedef __attribute__((ext_vector_type(4)))  uint32_t u32x4;
typedef __attribute__((ext_vector_type(8)))  uint32_t u32x8;

#define B_TOT   8192
#define DDIM    64
#define SLOTS   32
#define CODE    128
#define ANCH    32
#define LDC     4736   // combined row stride
#define BETA    0.25f

static __device__ inline v8f wmma_f16(v16h a, v16h b, v8f c) {
  return __builtin_amdgcn_wmma_f32_16x16x32_f16(false, a, false, b, (short)0, c, false, false);
}
// native f32 WMMA, K=4 (exact arithmetic for the VQ argmin path)
static __device__ inline v8f wmma_f32k4(v2f a, v2f b, v8f c) {
  return __builtin_amdgcn_wmma_f32_16x16x4_f32(false, a, false, b, (short)0, c, false, false);
}

// A-fragment (16x32 f16) from LDS f32 tile, row-major with leading dim ld.
static __device__ inline v16h load_A(const float* S, int ld, int k0, float scale) {
  const int lane = threadIdx.x & 31;
  const int m = lane & 15, g = lane >> 4;
  v16h a;
#pragma unroll
  for (int e = 0; e < 16; ++e) {
    const int k = 16 * g + e + k0;
    a[e] = (_Float16)(scale * S[m * ld + k]);
  }
  return a;
}

// B-fragment (32x16 f16) from global row-major W[k][n], leading dim ld.
static __device__ inline v16h load_B(const float* __restrict__ W, int ld, int k0, int n0) {
  const int lane = threadIdx.x & 31;
  const int n = n0 + (lane & 15);
  const int kb = k0 + ((lane >> 4) << 4);
  v16h b;
#pragma unroll
  for (int e = 0; e < 16; ++e) b[e] = (_Float16)W[(size_t)(kb + e) * ld + n];
  return b;
}

// B-fragment where B[k][n] = P[n][k]  (multiply by P^T), P row-major ld. Contiguous in e.
static __device__ inline v16h load_BT(const float* __restrict__ P, int ld, int k0, int n0) {
  const int lane = threadIdx.x & 31;
  const int n = n0 + (lane & 15);
  const int kb = k0 + ((lane >> 4) << 4);
  v16h b;
#pragma unroll
  for (int e = 0; e < 16; ++e) b[e] = (_Float16)P[(size_t)n * ld + kb + e];
  return b;
}

// B-fragment from LDS holding W^T ([n][k], ld=64). Contiguous 64B per lane -> ds_load_b128.
static __device__ inline v16h load_B_lt(const float* S, int k0, int n0) {
  const int lane = threadIdx.x & 31;
  const int n = n0 + (lane & 15);
  const int kb = k0 + ((lane >> 4) << 4);
  const int base = n * 64 + kb;
  v16h b;
#pragma unroll
  for (int e = 0; e < 16; ++e) b[e] = (_Float16)S[base + e];
  return b;
}

// stage a 64x64 row-major global matrix into LDS transposed ([n][k]); coalesced b128 reads
static __device__ inline void stage_WT(const float* __restrict__ W, float* S) {
  const int lane = threadIdx.x & 31;
#pragma unroll 4
  for (int t = lane; t < 1024; t += 32) {
    const float4 v = ((const float4*)W)[t];
    const int k = t >> 4;          // row in W
    const int n = (t & 15) * 4;    // col in W
    S[(n + 0) * 64 + k] = v.x;
    S[(n + 1) * 64 + k] = v.y;
    S[(n + 2) * 64 + k] = v.z;
    S[(n + 3) * 64 + k] = v.w;
  }
}

// store C/D accumulator tile to LDS f32 row-major tile (ld), cols n0..n0+15
static __device__ inline void store_D(float* S, int ld, int n0, v8f acc) {
  const int lane = threadIdx.x & 31;
  const int n = n0 + (lane & 15), g = lane >> 4;
#pragma unroll
  for (int r = 0; r < 8; ++r) S[(r + 8 * g) * ld + n] = acc[r];
}

// complex linear on a 16-row tile with LDS-staged transposed weights
template <typename F>
static __device__ inline void clinear16(const float* __restrict__ Wr, const float* __restrict__ Wi,
                                        const float* __restrict__ br, const float* __restrict__ bi,
                                        float* SWrT, float* SWiT,
                                        const v16h* ar, const v16h* ai, const v16h* ain, F sink) {
  const int lane = threadIdx.x & 31;
  __syncthreads();
  stage_WT(Wr, SWrT);
  stage_WT(Wi, SWiT);
  __syncthreads();
#pragma unroll
  for (int nt = 0; nt < 4; ++nt) {
    v8f aR = {}; v8f aI = {};
#pragma unroll
    for (int kt = 0; kt < 2; ++kt) {
      v16h bR = load_B_lt(SWrT, kt * 32, nt * 16);
      v16h bI = load_B_lt(SWiT, kt * 32, nt * 16);
      aR = wmma_f16(ar[kt],  bR, aR);
      aR = wmma_f16(ain[kt], bI, aR);   // - zi*Wi
      aI = wmma_f16(ar[kt],  bI, aI);
      aI = wmma_f16(ai[kt],  bR, aI);   // + zi*Wr
    }
    const float vbr = br[nt * 16 + (lane & 15)];
    const float vbi = bi[nt * 16 + (lane & 15)];
#pragma unroll
    for (int r = 0; r < 8; ++r) { aR[r] += vbr; aI[r] += vbi; }
    sink(nt, aR, aI);
  }
}

// ---------------- kernel 0: codebook norms + zero accumulators ----------------
__global__ __launch_bounds__(128) void k_prep(const float* __restrict__ codebook,
                                              float* cnorm, float* counts, float* slotent) {
  const int c = threadIdx.x;
  float s = 0.f;
  for (int k = 0; k < 128; ++k) { float v = codebook[c * 128 + k]; s += v * v; }
  cnorm[c] = s;
  counts[c] = 0.f;
  if (c == 0) slotent[0] = 0.f;
}

// ---------------- kernel 1: WMMA tile kernel (16 rows / wave) ----------------
__global__ __launch_bounds__(32) void k_main(
    const float* __restrict__ curr_r, const float* __restrict__ curr_i,
    const float* __restrict__ codebook,
    const float* __restrict__ qkv_Wr, const float* __restrict__ qkv_Wi,
    const float* __restrict__ qkv_br, const float* __restrict__ qkv_bi,
    const float* __restrict__ quad_Wr, const float* __restrict__ quad_Wi,
    const float* __restrict__ quad_br, const float* __restrict__ quad_bi,
    const float* __restrict__ sens_Wr, const float* __restrict__ sens_Wi,
    const float* __restrict__ sens_br, const float* __restrict__ sens_bi,
    const float* __restrict__ vis_pal, const float* __restrict__ aud_pal,
    const float* __restrict__ gate_W, const float* __restrict__ gate_b,
    const float* __restrict__ addr_W, const float* __restrict__ addr_b,
    const float* __restrict__ cnorm, float* counts, float* slotent,
    float* __restrict__ eff_ws, float* __restrict__ combined,
    float* __restrict__ vq_loss_out, float* __restrict__ idx_out) {
  const int lane = threadIdx.x;
  const int row0 = blockIdx.x * 16;

  __shared__ float SM[6144];       // Sr,Si,T0,T1,T2,T3 (each 16x64)
  __shared__ float SWT[8192];      // two 64x64 transposed weight tiles
  __shared__ float Sscore[512];    // 16x32 scratch (scores / attn / logits)
  __shared__ float znormS[16], gateS[16], cnormS[128];
  __shared__ int   idxS[16];

  float* Sr = SM;          float* Si = SM + 1024;
  float* T0 = SM + 2048;   float* T1 = SM + 3072;
  float* T2 = SM + 4096;   float* T3 = SM + 5120;
  float* SWrT = SWT;       float* SWiT = SWT + 4096;

  for (int i = lane; i < 1024; i += 32) {
    Sr[i] = curr_r[(size_t)row0 * 64 + i];
    Si[i] = curr_i[(size_t)row0 * 64 + i];
  }
  for (int i = lane; i < 128; i += 32) cnormS[i] = cnorm[i];
  __syncthreads();

  v16h ar[2], ai[2], ain[2];
  ar[0]  = load_A(Sr, 64, 0, 1.f);  ar[1]  = load_A(Sr, 64, 32, 1.f);
  ai[0]  = load_A(Si, 64, 0, 1.f);  ai[1]  = load_A(Si, 64, 32, 1.f);
  ain[0] = load_A(Si, 64, 0, -1.f); ain[1] = load_A(Si, 64, 32, -1.f);

  if (lane < 16) {
    float s = 0.f;
    for (int k = 0; k < 64; ++k) {
      float a = Sr[lane * 64 + k], b = Si[lane * 64 + k];
      s += a * a + b * b;
    }
    znormS[lane] = s;
  }
  __syncthreads();

  // ---- VQ in exact f32 via V_WMMA_F32_16X16X4_F32: d = |z|^2+|c|^2-2 z.c, argmin ----
  const int m_ = lane & 15, g_ = lane >> 4;
  float bestd[8]; int bestc[8];
#pragma unroll
  for (int r = 0; r < 8; ++r) { bestd[r] = 3.0e38f; bestc[r] = 0; }
  for (int nt = 0; nt < 8; ++nt) {
    v8f acc = {};
    const int c = nt * 16 + m_;
#pragma unroll
    for (int kq = 0; kq < 32; ++kq) {
      const int ka = kq * 4 + 2 * g_;           // K index of element 0 (same 64-half as +1)
      const float* Z = (ka < 64) ? Sr : Si;
      const int kk = ka & 63;
      v2f av = { Z[m_ * 64 + kk], Z[m_ * 64 + kk + 1] };
      v2f bv = { codebook[c * 128 + ka], codebook[c * 128 + ka + 1] };
      acc = wmma_f32k4(av, bv, acc);
    }
    const float cn = cnormS[c];
#pragma unroll
    for (int r = 0; r < 8; ++r) {
      float dd = znormS[r + 8 * g_] + cn - 2.f * acc[r];
      dd = fmaxf(dd, 0.f);
      if (dd < bestd[r]) { bestd[r] = dd; bestc[r] = c; }
    }
  }
#pragma unroll
  for (int mask = 1; mask < 16; mask <<= 1) {
#pragma unroll
    for (int r = 0; r < 8; ++r) {
      float od = __shfl_xor(bestd[r], mask, 32);
      int   oc = __shfl_xor(bestc[r], mask, 32);
      if (od < bestd[r] || (od == bestd[r] && oc < bestc[r])) { bestd[r] = od; bestc[r] = oc; }
    }
  }
  if ((lane & 15) == 0) {
#pragma unroll
    for (int r = 0; r < 8; ++r) idxS[r + 8 * (lane >> 4)] = bestc[r];
  }
  __syncthreads();

  if (lane < 16) {
    const int m = lane, c = idxS[m];
    float s = 0.f;
    for (int k = 0; k < 128; ++k) {
      const float z = (k < 64) ? Sr[m * 64 + k] : Si[m * 64 + k - 64];
      const float d = codebook[c * 128 + k] - z;
      s += d * d;
    }
    vq_loss_out[row0 + m] = BETA * s * (1.f / 128.f);
    idx_out[row0 + m] = (float)c;
    atomicAdd(&counts[c], 1.0f);
  }
  // straight-through value: z_q = codebook[idx]  -> combined cols [0,128)
  for (int t = lane; t < 2048; t += 32) {
    const int m = t >> 7, k = t & 127;
    combined[(size_t)(row0 + m) * LDC + k] = codebook[idxS[m] * 128 + k];
  }

  // f16 A-fragments of z_flat for the addr-logits GEMM
  v16h zf[4] = { ar[0], ar[1], ai[0], ai[1] };

  // ---- q, k linears -> LDS, gate ----
  clinear16(qkv_Wr, qkv_Wi, qkv_br, qkv_bi, SWrT, SWiT, ar, ai, ain,
            [&](int nt, v8f aR, v8f aI) { store_D(T0, 64, nt * 16, aR); store_D(T1, 64, nt * 16, aI); });
  clinear16(qkv_Wr + 4096, qkv_Wi + 4096, qkv_br + 64, qkv_bi + 64, SWrT, SWiT, ar, ai, ain,
            [&](int nt, v8f aR, v8f aI) { store_D(T2, 64, nt * 16, aR); store_D(T3, 64, nt * 16, aI); });
  __syncthreads();
  if (lane < 16) {
    float g = 0.f;
    for (int k = 0; k < 64; ++k)
      g += T0[lane * 64 + k] * T2[lane * 64 + k] + T1[lane * 64 + k] * T3[lane * 64 + k];
    gateS[lane] = 1.f / (1.f + expf(-g));
  }
  __syncthreads();

  // ---- v linear -> g_r,g_i = v*gate -> combined cols [128,256) ----
  clinear16(qkv_Wr + 8192, qkv_Wi + 8192, qkv_br + 128, qkv_bi + 128, SWrT, SWiT, ar, ai, ain,
            [&](int nt, v8f aR, v8f aI) {
              const int n = nt * 16 + (lane & 15), g = lane >> 4;
#pragma unroll
              for (int r = 0; r < 8; ++r) {
                const int m = r + 8 * g; const float gt = gateS[m];
                combined[(size_t)(row0 + m) * LDC + 128 + n] = aR[r] * gt;
                combined[(size_t)(row0 + m) * LDC + 192 + n] = aI[r] * gt;
              }
            });

  // ---- quad linear: q_r=-zi, q_i=zr -> combined cols [512,640) ----
  clinear16(quad_Wr, quad_Wi, quad_br, quad_bi, SWrT, SWiT, ar, ai, ain,
            [&](int nt, v8f aR, v8f aI) {
              const int n = nt * 16 + (lane & 15), g = lane >> 4;
#pragma unroll
              for (int r = 0; r < 8; ++r) {
                const int m = r + 8 * g;
                combined[(size_t)(row0 + m) * LDC + 512 + n] = -aI[r];
                combined[(size_t)(row0 + m) * LDC + 576 + n] = aR[r];
              }
            });

  // ---- sens linear -> T0/T1 (sflat) ----
  clinear16(sens_Wr, sens_Wi, sens_br, sens_bi, SWrT, SWiT, ar, ai, ain,
            [&](int nt, v8f aR, v8f aI) { store_D(T0, 64, nt * 16, aR); store_D(T1, 64, nt * 16, aI); });
  __syncthreads();
  v16h sf[4];
  sf[0] = load_A(T0, 64, 0, 1.f);  sf[1] = load_A(T0, 64, 32, 1.f);
  sf[2] = load_A(T1, 64, 0, 1.f);  sf[3] = load_A(T1, 64, 32, 1.f);

  // ---- palettes: exp_r = vis[:, :D] - aud[:, D:], exp_i = vis[:, D:] + aud[:, :D] ----
  float* R = T2;  // 16 x 128 recon buffer (spans T2+T3)
#pragma unroll
  for (int p = 0; p < 2; ++p) {
    const float* pal = p ? aud_pal : vis_pal;
#pragma unroll
    for (int nt = 0; nt < 2; ++nt) {
      v8f acc = {};
#pragma unroll
      for (int kt = 0; kt < 4; ++kt) {
        v16h bf = load_BT(pal, 128, kt * 32, nt * 16);
        acc = wmma_f16(sf[kt], bf, acc);
      }
      store_D(Sscore, 32, nt * 16, acc);
    }
    __syncthreads();
    if (lane < 16) {   // rowwise softmax over 32 anchors, in place
      const int m = lane;
      float mx = -3.0e38f;
      for (int s = 0; s < 32; ++s) mx = fmaxf(mx, Sscore[m * 32 + s]);
      float den = 0.f;
      for (int s = 0; s < 32; ++s) { float e = expf(Sscore[m * 32 + s] - mx); Sscore[m * 32 + s] = e; den += e; }
      const float inv = 1.f / den;
      for (int s = 0; s < 32; ++s) Sscore[m * 32 + s] *= inv;
    }
    __syncthreads();
    v16h af = load_A(Sscore, 32, 0, 1.f);   // attn weights as K=32 A fragment
#pragma unroll
    for (int nt = 0; nt < 8; ++nt) {
      v8f acc = {};
      v16h bf = load_B(pal, 128, 0, nt * 16);
      acc = wmma_f16(af, bf, acc);
      const int n = nt * 16 + (lane & 15), g = lane >> 4;
      if (p == 0) {
#pragma unroll
        for (int r = 0; r < 8; ++r) R[(r + 8 * g) * 128 + n] = acc[r];
      } else {
#pragma unroll
        for (int r = 0; r < 8; ++r) {
          const int m = r + 8 * g;
          if (nt < 4) R[m * 128 + 64 + n] += acc[r];   // exp_i += aud[:, :D]
          else        R[m * 128 + (n - 64)] -= acc[r]; // exp_r -= aud[:, D:]
        }
      }
    }
    __syncthreads();
  }
  for (int t = lane; t < 2048; t += 32) {
    const int m = t >> 7, n = t & 127;
    combined[(size_t)(row0 + m) * LDC + 384 + n] = R[t];
  }

  // ---- addressing: logits -> softmax -> entropy, top-3, eff = wg * sparse ----
#pragma unroll
  for (int nt = 0; nt < 2; ++nt) {
    v8f acc = {};
#pragma unroll
    for (int kt = 0; kt < 4; ++kt) {
      v16h bf = load_B(addr_W, 32, kt * 32, nt * 16);
      acc = wmma_f16(zf[kt], bf, acc);
    }
    const float ab = addr_b[nt * 16 + (lane & 15)];
#pragma unroll
    for (int r = 0; r < 8; ++r) acc[r] += ab;
    store_D(Sscore, 32, nt * 16, acc);
  }
  __syncthreads();
  if (lane < 16) {
    const int m = lane;
    float pb[32];
    float mx = -3.0e38f;
    for (int s = 0; s < 32; ++s) mx = fmaxf(mx, Sscore[m * 32 + s]);
    float den = 0.f;
    for (int s = 0; s < 32; ++s) { float e = expf(Sscore[m * 32 + s] - mx); pb[s] = e; den += e; }
    float ent = 0.f;
    const float invden = 1.f / den;
    for (int s = 0; s < 32; ++s) { pb[s] *= invden; ent -= pb[s] * logf(pb[s] + 1e-10f); }
    atomicAdd(slotent, ent);
    float wa = gate_b[0];
    for (int k = 0; k < 128; ++k) {
      const float z = (k < 64) ? Sr[m * 64 + k] : Si[m * 64 + k - 64];
      wa += z * gate_W[k];
    }
    const float wg = 1.f / (1.f + expf(-wa));
    int t0 = -1, t1 = -1, t2 = -1; float v0 = -1.f, v1 = -1.f, v2 = -1.f;
    for (int s = 0; s < 32; ++s) {
      const float v = pb[s];
      if (v > v0)      { v2 = v1; t2 = t1; v1 = v0; t1 = t0; v0 = v; t0 = s; }
      else if (v > v1) { v2 = v1; t2 = t1; v1 = v;  t1 = s; }
      else if (v > v2) { v2 = v;  t2 = s; }
    }
    const float inv = 1.f / (v0 + v1 + v2 + 1e-6f);
    for (int s = 0; s < 32; ++s) {
      const float sp = (s == t0) ? v0 : (s == t1) ? v1 : (s == t2) ? v2 : 0.f;
      eff_ws[(size_t)(row0 + m) * 32 + s] = wg * sp * inv;
    }
  }
}

// ---------------- kernel 2: streaming memory kernel (1 row / block) ----------------
// mem tile (2 x 8KB) pulled into LDS by the Tensor Data Mover, not VGPR bounce.
__global__ __launch_bounds__(256) void k_mem(
    const float* __restrict__ mem_r, const float* __restrict__ mem_i,
    const float* __restrict__ curr_r, const float* __restrict__ curr_i,
    const float* __restrict__ eff_ws,
    const float* __restrict__ ln_gr, const float* __restrict__ ln_br,
    const float* __restrict__ ln_gi, const float* __restrict__ ln_bi,
    float* __restrict__ combined) {
  const int b = blockIdx.x, tid = threadIdx.x;
  __shared__ float Mr[2048], Mi[2048], Cr[64], Ci[64];
  __shared__ float simS[32], attnS[32], effS[32];
  __shared__ float Gr[64], Br[64], Gi[64], Bi[64];

  if (tid < 32) {   // wave 0 issues two TDM descriptors (EXEC-independent, TENSORcnt-tracked)
    const uint64_t gar = (uint64_t)(uintptr_t)(mem_r + (size_t)b * 2048);
    const uint64_t gai = (uint64_t)(uintptr_t)(mem_i + (size_t)b * 2048);
    const uint32_t ldsr = (uint32_t)(uintptr_t)(&Mr[0]);
    const uint32_t ldsi = (uint32_t)(uintptr_t)(&Mi[0]);
    // D# group0: count=1 | lds_addr | global_addr | type=2 ("image")
    u32x4 g0r = { 1u, ldsr, (uint32_t)gar, (uint32_t)(gar >> 32) | (2u << 30) };
    u32x4 g0i = { 1u, ldsi, (uint32_t)gai, (uint32_t)(gai >> 32) | (2u << 30) };
    // D# group1: data_size=4B; tensor 2048x1; tile 2048x1; dim0 stride 2048
    u32x8 g1 = { (2u << 16),          // wg_mask=0 | data_size=2 (4B)
                 (2048u << 16),       // atomic_bar=0 | tensor_dim0[15:0]=2048
                 (1u << 16),          // tensor_dim0[31:16]=0 | tensor_dim1[15:0]=1
                 (2048u << 16),       // tensor_dim1[31:16]=0 | tile_dim0=2048
                 1u,                  // tile_dim1=1 | tile_dim2=0
                 2048u,               // tensor_dim0_stride[31:0]
                 (2048u << 16),       // stride0[47:32]=0 | tensor_dim1_stride[15:0]=2048
                 0u };
    asm volatile("tensor_load_to_lds %0, %1" :: "s"(g0r), "s"(g1) : "memory");
    asm volatile("tensor_load_to_lds %0, %1" :: "s"(g0i), "s"(g1) : "memory");
    __builtin_amdgcn_s_wait_tensorcnt(0);
  }
  if (tid < 64) {
    Cr[tid] = curr_r[(size_t)b * 64 + tid];  Ci[tid] = curr_i[(size_t)b * 64 + tid];
    Gr[tid] = ln_gr[tid]; Br[tid] = ln_br[tid]; Gi[tid] = ln_gi[tid]; Bi[tid] = ln_bi[tid];
  }
  if (tid >= 64 && tid < 96) effS[tid - 64] = eff_ws[(size_t)b * 32 + (tid - 64)];
  __syncthreads();

  const int slot = tid >> 3, l = tid & 7;
  // sim[s] = mem_r[s].curr_r + mem_i[s].curr_i  (8 lanes per slot)
  float part = 0.f;
#pragma unroll
  for (int j = 0; j < 8; ++j) {
    const int d = l * 8 + j;
    part += Mr[slot * 64 + d] * Cr[d] + Mi[slot * 64 + d] * Ci[d];
  }
  part += __shfl_xor(part, 1, 32);
  part += __shfl_xor(part, 2, 32);
  part += __shfl_xor(part, 4, 32);
  if (l == 0) simS[slot] = part;
  __syncthreads();

  {
    float mx = -3.0e38f;
    for (int s = 0; s < 32; ++s) mx = fmaxf(mx, simS[s]);
    float den = 0.f;
    for (int s = 0; s < 32; ++s) den += expf(simS[s] - mx);
    if (tid < 32) attnS[tid] = expf(simS[tid] - mx) / den;
  }
  __syncthreads();

  // read_r / read_i -> combined cols [256,384)
  if (tid < 128) {
    const int half = tid >> 6, d = tid & 63;
    const float* M = half ? Mi : Mr;
    float a = 0.f;
    for (int s = 0; s < 32; ++s) a += attnS[s] * M[s * 64 + d];
    combined[(size_t)b * LDC + 256 + tid] = a;
  }

  // nm = layernorm(tanh(mem + eff*(curr - mem))) -> cols [640,2688) and [2688,4736)
  const float e = effS[slot];
#pragma unroll
  for (int half = 0; half < 2; ++half) {
    const float* M = half ? Mi : Mr;
    const float* C = half ? Ci : Cr;
    const float* Gm = half ? Gi : Gr;
    const float* Bt = half ? Bi : Br;
    float tv[8], sm = 0.f, sq = 0.f;
#pragma unroll
    for (int j = 0; j < 8; ++j) {
      const int d = l * 8 + j;
      const float x = M[slot * 64 + d];
      const float t = tanhf(x + e * (C[d] - x));
      tv[j] = t; sm += t; sq += t * t;
    }
    sm += __shfl_xor(sm, 1, 32); sm += __shfl_xor(sm, 2, 32); sm += __shfl_xor(sm, 4, 32);
    sq += __shfl_xor(sq, 1, 32); sq += __shfl_xor(sq, 2, 32); sq += __shfl_xor(sq, 4, 32);
    const float mean = sm * (1.f / 64.f);
    const float var  = sq * (1.f / 64.f) - mean * mean;
    const float inv  = rsqrtf(var + 1e-6f);
    const size_t base = (size_t)b * LDC + 640 + (size_t)half * 2048 + slot * 64;
#pragma unroll
    for (int j = 0; j < 8; ++j) {
      const int d = l * 8 + j;
      combined[base + d] = (tv[j] - mean) * inv * Gm[d] + Bt[d];
    }
  }
}

// ---------------- kernel 3: scalar outputs ----------------
__global__ __launch_bounds__(128) void k_final(const float* __restrict__ counts,
                                               const float* __restrict__ slotent,
                                               float* __restrict__ scal) {
  __shared__ float red[128];
  const int t = threadIdx.x;
  const float p = counts[t] * (1.f / (float)B_TOT);
  red[t] = -p * logf(p + 1e-10f);
  __syncthreads();
  for (int s = 64; s > 0; s >>= 1) {
    if (t < s) red[t] += red[t + s];
    __syncthreads();
  }
  if (t == 0) {
    scal[0] = slotent[0] * (1.f / (float)B_TOT);       // slot_entropy
    scal[1] = red[0] / logf(128.f);                    // norm_entropy
  }
}

extern "C" void kernel_launch(void* const* d_in, const int* in_sizes, int n_in,
                              void* d_out, int out_size, void* d_ws, size_t ws_size,
                              hipStream_t stream) {
  const float* curr_r   = (const float*)d_in[0];
  const float* curr_i   = (const float*)d_in[1];
  const float* mem_r    = (const float*)d_in[2];
  const float* mem_i    = (const float*)d_in[3];
  const float* codebook = (const float*)d_in[4];
  const float* qkv_Wr   = (const float*)d_in[5];
  const float* qkv_Wi   = (const float*)d_in[6];
  const float* qkv_br   = (const float*)d_in[7];
  const float* qkv_bi   = (const float*)d_in[8];
  const float* quad_Wr  = (const float*)d_in[9];
  const float* quad_Wi  = (const float*)d_in[10];
  const float* quad_br  = (const float*)d_in[11];
  const float* quad_bi  = (const float*)d_in[12];
  const float* sens_Wr  = (const float*)d_in[13];
  const float* sens_Wi  = (const float*)d_in[14];
  const float* sens_br  = (const float*)d_in[15];
  const float* sens_bi  = (const float*)d_in[16];
  const float* vis_pal  = (const float*)d_in[17];
  const float* aud_pal  = (const float*)d_in[18];
  const float* gate_W   = (const float*)d_in[19];
  const float* gate_b   = (const float*)d_in[20];
  const float* addr_W   = (const float*)d_in[21];
  const float* addr_b   = (const float*)d_in[22];
  const float* ln_gr    = (const float*)d_in[23];
  const float* ln_br    = (const float*)d_in[24];
  const float* ln_gi    = (const float*)d_in[25];
  const float* ln_bi    = (const float*)d_in[26];

  float* out       = (float*)d_out;
  float* combined  = out;
  float* vq_loss   = out + (size_t)B_TOT * LDC;
  float* idx_out   = vq_loss + B_TOT;
  float* scal      = idx_out + B_TOT;

  float* cnorm   = (float*)d_ws;            // 128
  float* counts  = cnorm + 128;             // 128
  float* slotent = counts + 128;            // 1
  float* eff_ws  = (float*)((char*)d_ws + 4096);  // B*32 floats

  k_prep<<<1, 128, 0, stream>>>(codebook, cnorm, counts, slotent);
  k_main<<<B_TOT / 16, 32, 0, stream>>>(
      curr_r, curr_i, codebook,
      qkv_Wr, qkv_Wi, qkv_br, qkv_bi,
      quad_Wr, quad_Wi, quad_br, quad_bi,
      sens_Wr, sens_Wi, sens_br, sens_bi,
      vis_pal, aud_pal, gate_W, gate_b, addr_W, addr_b,
      cnorm, counts, slotent, eff_ws, combined, vq_loss, idx_out);
  k_mem<<<B_TOT, 256, 0, stream>>>(mem_r, mem_i, curr_r, curr_i, eff_ws,
                                   ln_gr, ln_br, ln_gi, ln_bi, combined);
  k_final<<<1, 128, 0, stream>>>(counts, slotent, scal);
}
